// GPTQLinear_11630771438119
// MI455X (gfx1250) — matrix operand order repaired and checked by reading, not compiled, not run on hardware
//
#include <hip/hip_runtime.h>

typedef __bf16 bf16;
typedef __attribute__((ext_vector_type(4)))  __bf16 v4bf;
typedef __attribute__((ext_vector_type(8)))  __bf16 v8bf;
typedef __attribute__((ext_vector_type(16))) __bf16 v16bf;
typedef __attribute__((ext_vector_type(4)))  float  v4f;
typedef __attribute__((ext_vector_type(8)))  float  v8f;

#define M_DIM   8192            // B*S = 4*2048
#define K_DIM   4096
#define N_DIM   16384
#define KW      (K_DIM / 8)     // qweight words per row = 512
#define GROUPS  32              // K / 128
#define GZW     (GROUPS / 8)    // qzeros words per row = 4

#define BM 128
#define BN 128
#define BK 64
#define LDSS 72                 // padded LDS row stride in bf16 (multiple of 8, != 64)

__global__ __launch_bounds__(256)
void gptq_gemm_bf16_wmma(const float* __restrict__ x,
                         const int*   __restrict__ qweight,
                         const int*   __restrict__ qzeros,
                         const float* __restrict__ scales,
                         const float* __restrict__ bias,
                         float* __restrict__ out)
{
    __shared__ bf16 Xs[BM * LDSS];   // 18432 B
    __shared__ bf16 Ws[BN * LDSS];   // 18432 B

    const int tid   = threadIdx.x;
    const int lane  = tid & 31;
    const int wave  = tid >> 5;      // 0..7
    const int waveM = wave & 1;      // 2 wave-rows
    const int waveN = wave >> 1;     // 4 wave-cols
    const int r     = lane & 15;     // 0..15
    const int h     = lane >> 4;     // 0..1

    const int gm0 = blockIdx.y * BM;
    const int gn0 = blockIdx.x * BN;
    const int wm0 = waveM * 64;      // wave sub-tile: 64 M x 32 N
    const int wn0 = waveN * 32;

    v8f acc[4][2];
    #pragma unroll
    for (int i = 0; i < 4; ++i)
        #pragma unroll
        for (int j = 0; j < 2; ++j)
            acc[i][j] = (v8f){};

    for (int kt = 0; kt < K_DIM / BK; ++kt) {
        __syncthreads();   // previous iteration's LDS reads done

        // ---- stage X tile: 128 x 64 fp32 -> bf16 into LDS ----
        {
            const int kbase = kt * BK;
            #pragma unroll
            for (int t = 0; t < 8; ++t) {
                const int id  = tid + t * 256;       // 0..2047 float4 slots
                const int row = id >> 4;             // /16 float4 per row
                const int c4  = (id & 15) << 2;      // column (floats)
                const v4f xv = *(const v4f*)(x + (size_t)(gm0 + row) * K_DIM + kbase + c4);
                v4bf b;
                b[0] = (bf16)xv[0]; b[1] = (bf16)xv[1];
                b[2] = (bf16)xv[2]; b[3] = (bf16)xv[3];
                *(v4bf*)(&Xs[row * LDSS + c4]) = b;
            }
        }

        // ---- stage W tile: dequant 128 rows x 64 K nibbles -> bf16 into LDS ----
        {
            const int g      = kt >> 1;              // group index (BK=64, group=128)
            const int gword  = g >> 3;
            const int zshift = (g & 7) * 4;
            #pragma unroll
            for (int t = 0; t < 4; ++t) {
                const int id = tid + t * 256;        // 0..1023 qweight words
                const int n  = id >> 3;
                const int w  = id & 7;               // word within BK (8 nibbles each)
                const int gn = gn0 + n;
                const int word  = qweight[(size_t)gn * KW + kt * 8 + w];
                const int zword = qzeros[(size_t)gn * GZW + gword];
                const float z = (float)((zword >> zshift) & 0xF);
                const float s = scales[(size_t)gn * GROUPS + g];
                v8bf wv;
                #pragma unroll
                for (int j = 0; j < 8; ++j) {
                    const float q = (float)((word >> (4 * j)) & 0xF);
                    wv[j] = (bf16)((q - z) * s);
                }
                *(v8bf*)(&Ws[n * LDSS + w * 8]) = wv;
            }
        }

        __syncthreads();

        // ---- WMMA over BK in two K=32 steps ----
        #pragma unroll
        for (int kk = 0; kk < 2; ++kk) {
            const int k0 = kk * 32;
            v16bf a[4], b[2];

            // A fragment (16x32 bf16): halves 0-7 = K h*8+0..7, halves 8-15 = K 16+h*8+0..7
            #pragma unroll
            for (int i = 0; i < 4; ++i) {
                const bf16* p = &Xs[(wm0 + i * 16 + r) * LDSS + k0 + h * 8];
                const v8bf lo = *(const v8bf*)(p);
                const v8bf hi = *(const v8bf*)(p + 16);
                a[i] = __builtin_shufflevector(lo, hi,
                        0, 1, 2, 3, 4, 5, 6, 7, 8, 9, 10, 11, 12, 13, 14, 15);
            }

            // B fragment (32x16 bf16): col = lane%16, halves 0-7 = K h*8+0..7, halves 8-15 = K 16+h*8+0..7
            #pragma unroll
            for (int j = 0; j < 2; ++j) {
                const bf16* p = &Ws[(wn0 + j * 16 + r) * LDSS + k0 + h * 8];
                const v8bf lo = *(const v8bf*)(p);
                const v8bf hi = *(const v8bf*)(p + 16);
                b[j] = __builtin_shufflevector(lo, hi,
                        0, 1, 2, 3, 4, 5, 6, 7, 8, 9, 10, 11, 12, 13, 14, 15);
            }

            #pragma unroll
            for (int i = 0; i < 4; ++i)
                #pragma unroll
                for (int j = 0; j < 2; ++j)
                    acc[i][j] = __builtin_amdgcn_wmma_f32_16x16x32_bf16(
                        /*neg_a=*/false, a[i], /*neg_b=*/false, b[j],
                        /*c_mod=*/(short)0, acc[i][j],
                        /*reuse_a=*/false, /*reuse_b=*/false);
        }
    }

    // ---- epilogue: D layout is M = h*8 + v, N = lane%16 per fragment ----
    #pragma unroll
    for (int j = 0; j < 2; ++j) {
        const int   ncol = gn0 + wn0 + j * 16 + r;
        const float bv   = bias[ncol];
        #pragma unroll
        for (int i = 0; i < 4; ++i) {
            #pragma unroll
            for (int v = 0; v < 8; ++v) {
                const int mrow = gm0 + wm0 + i * 16 + h * 8 + v;
                out[(size_t)mrow * N_DIM + ncol] = acc[i][j][v] + bv;
            }
        }
    }
}

extern "C" void kernel_launch(void* const* d_in, const int* in_sizes, int n_in,
                              void* d_out, int out_size, void* d_ws, size_t ws_size,
                              hipStream_t stream)
{
    const float* x       = (const float*)d_in[0];
    const int*   qweight = (const int*)  d_in[1];
    const int*   qzeros  = (const int*)  d_in[2];
    const float* scales  = (const float*)d_in[3];
    const float* bias    = (const float*)d_in[4];
    float*       out     = (float*)d_out;

    dim3 grid(N_DIM / BN, M_DIM / BM);   // 128 x 64 blocks
    dim3 block(256);                     // 8 wave32s
    hipLaunchKernelGGL(gptq_gemm_bf16_wmma, grid, block, 0, stream,
                       x, qweight, qzeros, scales, bias, out);
}
